// model_v6_3221225472160
// MI455X (gfx1250) — matrix operand order repaired and checked by reading, not compiled, not run on hardware
//
#include <hip/hip_runtime.h>
#include <hip/hip_bf16.h>
#include <math.h>

// ---------------- model constants ----------------
#define Bc   32
#define Tc   512
#define Dc   256
#define Hc   8
#define HSc  8
#define NLc  4
#define DEPc 3
#define KSc  7
#define PREDc 64

// ---------------- WMMA types ----------------
typedef __attribute__((ext_vector_type(16))) __bf16        v16bf;
typedef __attribute__((ext_vector_type(8)))  float         v8f;
typedef __attribute__((ext_vector_type(4)))  unsigned int  u32x4;

union Frag {
  v16bf v;
  unsigned short u[16];
  unsigned int   w[8];
  u32x4          q[2];
};

__device__ __forceinline__ unsigned short bf16bits(float f) {
  union { float f; unsigned u; } c; c.f = f;
  unsigned r = c.u + 0x7FFFu + ((c.u >> 16) & 1u);   // round-to-nearest-even
  return (unsigned short)(r >> 16);
}
__device__ __forceinline__ unsigned int bfpack(float lo, float hi) {
  return (unsigned int)bf16bits(lo) | ((unsigned int)bf16bits(hi) << 16);
}
__device__ __forceinline__ float gelu_exact(float v) {
  return 0.5f * v * (1.0f + erff(v * 0.70710678118654752f));
}

// =====================================================================
// bf16-in / f32-acc WMMA GEMM:  C = act(A @ W + bias) (+ resid)
// Block: 128 threads (4 waves). Tile: 16(M) x 128(N), K-step 32.
// Each wave owns two 16x16 accumulators sharing one A fragment.
// A tile staged row-major As[m][k]; W tile staged TRANSPOSED Wt[n][k]
// so both fragment halves are contiguous 16B runs -> ds_load_b128.
// GUARD=false: all tiles full (no bounds checks anywhere).
// act: 0 none, 1 exact GELU, 2 sigmoid
// =====================================================================
template <bool GUARD>
__global__ void wmma_gemm_kernel(const float* __restrict__ A, int lda, long long sA,
                                 const float* __restrict__ W,
                                 const float* __restrict__ bias,
                                 const float* __restrict__ resid, int ldr, long long sR,
                                 float* __restrict__ C, int ldc, long long sC,
                                 int M, int N, int K, int act) {
  __shared__ unsigned short As[16][32];    // [m][k]
  __shared__ unsigned short Wt[128][32];   // [n][k]   (transposed W tile)

  const int b     = blockIdx.z;
  const float* Ab = A + (long long)b * sA;
  float*       Cb = C + (long long)b * sC;
  const float* Rb = resid ? (resid + (long long)b * sR) : nullptr;

  const int m0   = blockIdx.y * 16;
  const int nblk = blockIdx.x * 128;
  const int tid  = threadIdx.x;
  const int wave = tid >> 5;
  const int lane = tid & 31;

  const int row = lane & 15;
  const int akb = (lane < 16) ? 0 : 8;    // unused (A also loads at bkb)
  const int bkb = (lane < 16) ? 0 : 16;   // B k-base (ushorts)
  const int rb  = (lane >> 4) << 3;

  v8f acc0 = {};
  v8f acc1 = {};

  for (int k0 = 0; k0 < K; k0 += 32) {
    if (GUARD) {
      // scalar guarded staging (only used by tiny head GEMMs)
      for (int i = tid; i < 16 * 32; i += 128) {
        int r = i >> 5, c = i & 31;
        int gm = m0 + r;
        float v = (gm < M) ? Ab[(long long)gm * lda + (k0 + c)] : 0.0f;
        As[r][c] = bf16bits(v);
      }
      for (int i = tid; i < 32 * 128; i += 128) {
        int k = i >> 7, n = i & 127;
        int gn = nblk + n;
        float v = (gn < N) ? W[(long long)(k0 + k) * N + gn] : 0.0f;
        Wt[n][k] = bf16bits(v);
      }
    } else {
      // vectorized staging: A — one float4 per thread, packed b64 LDS store
      {
        const int r  = tid >> 3;
        const int c4 = (tid & 7) * 4;
        const float4 v = *(const float4*)&Ab[(long long)(m0 + r) * lda + (k0 + c4)];
        unsigned long long p = (unsigned long long)bfpack(v.x, v.y) |
                               ((unsigned long long)bfpack(v.z, v.w) << 32);
        *(unsigned long long*)&As[r][c4] = p;
      }
      // W — 8 float4 per thread, transposed b16 LDS stores
      #pragma unroll
      for (int it = 0; it < 8; ++it) {
        const int id = tid + it * 128;     // 0..1023
        const int k  = id >> 5;            // 0..31
        const int n4 = (id & 31) * 4;      // 0..124
        const float4 v = *(const float4*)&W[(long long)(k0 + k) * N + nblk + n4];
        Wt[n4 + 0][k] = bf16bits(v.x);
        Wt[n4 + 1][k] = bf16bits(v.y);
        Wt[n4 + 2][k] = bf16bits(v.z);
        Wt[n4 + 3][k] = bf16bits(v.w);
      }
      // prefetch next K tile (global_prefetch_b8)
      if (k0 + 32 < K) {
        __builtin_prefetch(&Ab[(long long)(m0 + (tid & 15)) * lda + k0 + 32], 0, 3);
        __builtin_prefetch(&W[(long long)(k0 + 32 + lane) * N + nblk], 0, 3);
      }
    }
    __syncthreads();

    // fragments: contiguous 16B halves -> ds_load_b128
    Frag af;
    af.q[0] = *(const u32x4*)&As[row][bkb];
    af.q[1] = *(const u32x4*)&As[row][bkb + 8];

    const int c0 = wave * 32 + (lane & 15);
    Frag bf0, bf1;
    bf0.q[0] = *(const u32x4*)&Wt[c0][bkb];
    bf0.q[1] = *(const u32x4*)&Wt[c0][bkb + 8];
    bf1.q[0] = *(const u32x4*)&Wt[c0 + 16][bkb];
    bf1.q[1] = *(const u32x4*)&Wt[c0 + 16][bkb + 8];

    acc0 = __builtin_amdgcn_wmma_f32_16x16x32_bf16(false, af.v, false, bf0.v,
                                                   (short)0, acc0, false, false);
    acc1 = __builtin_amdgcn_wmma_f32_16x16x32_bf16(false, af.v, false, bf1.v,
                                                   (short)0, acc1, false, false);
    __syncthreads();
  }

  // epilogue: bias + activation + optional residual (two 16-col groups)
  const int col = lane & 15;
  #pragma unroll
  for (int g = 0; g < 2; ++g) {
    const v8f& acc = g ? acc1 : acc0;
    const int gn = nblk + wave * 32 + g * 16 + col;
    if (!GUARD || gn < N) {
      const float bv = bias ? bias[gn] : 0.0f;
      #pragma unroll
      for (int r = 0; r < 8; ++r) {
        const int gm = m0 + rb + r;
        if (!GUARD || gm < M) {
          float v = acc[r] + bv;
          if (act == 1)      v = gelu_exact(v);
          else if (act == 2) v = 1.0f / (1.0f + __expf(-v));
          if (Rb) v += Rb[(long long)gm * ldr + gn];
          Cb[(long long)gm * ldc + gn] = v;
        }
      }
    }
  }
}

// =====================================================================
// Fused WMMA attention: O[b,n,h*hd+d] = softmax(Q Kt * scale + mask) V
// Q from QKVq slot 0; K,V from QKVkv slots 1,2 (cross-attn capable).
// 1 wave per 16-query tile; 16 x Ntok score tile lives in LDS.
// MASKED: dilated-causal mask; dilation is a power of two, so the test
// is pure bitwise (d & (dil-1)) / (d >> log2dil) -> branchless cndmask.
// =====================================================================
template <bool MASKED>
__global__ void wmma_attn_kernel(const float* __restrict__ QKVq,
                                 const float* __restrict__ QKVkv,
                                 float* __restrict__ O,
                                 int Ntok, int heads, int hd,
                                 float scale, int dil_m1, int dil_sh, int ks) {
  extern __shared__ float Slds[];     // 16 x Ntok score/prob tile
  __shared__ float rinv[16];

  const int q0 = blockIdx.x * 16;
  const int h  = blockIdx.y;
  const int b  = blockIdx.z;
  const int Dm = heads * hd;
  const int rs = 3 * Dm;
  const long long baseq = (long long)b * Ntok * rs;
  const float* Qp = QKVq  + baseq + h * hd;
  const float* Kp = QKVkv + baseq + Dm + h * hd;
  const float* Vp = QKVkv + baseq + 2 * Dm + h * hd;

  const int lane = threadIdx.x;
  const int row  = lane & 15;
  const int col  = lane & 15;
  const int akb  = (lane < 16) ? 0 : 8;
  const int bkb  = (lane < 16) ? 0 : 16;
  const int rb   = (lane >> 4) << 3;

  // ---- phase 1: S = Q K^T * scale + mask  -> LDS ----
  for (int j0 = 0; j0 < Ntok; j0 += 16) {
    v8f sacc = {};
    for (int d0 = 0; d0 < hd; d0 += 32) {
      Frag af, bfr;
      const float* qrow = Qp + (long long)(q0 + row) * rs + d0;
      const float4 qa = *(const float4*)(qrow + akb);
      const float4 qb = *(const float4*)(qrow + akb + 4);
      const float4 qc = *(const float4*)(qrow + akb + 16);
      const float4 qd = *(const float4*)(qrow + akb + 20);
      af.w[0] = bfpack(qa.x, qa.y); af.w[1] = bfpack(qa.z, qa.w);
      af.w[2] = bfpack(qb.x, qb.y); af.w[3] = bfpack(qb.z, qb.w);
      af.w[4] = bfpack(qc.x, qc.y); af.w[5] = bfpack(qc.z, qc.w);
      af.w[6] = bfpack(qd.x, qd.y); af.w[7] = bfpack(qd.z, qd.w);

      const float* krow = Kp + (long long)(j0 + col) * rs + d0 + akb;
      const float4 ka = *(const float4*)(krow);
      const float4 kb = *(const float4*)(krow + 4);
      const float4 kc = *(const float4*)(krow + 16);
      const float4 kd = *(const float4*)(krow + 20);
      bfr.w[0] = bfpack(ka.x, ka.y); bfr.w[1] = bfpack(ka.z, ka.w);
      bfr.w[2] = bfpack(kb.x, kb.y); bfr.w[3] = bfpack(kb.z, kb.w);
      bfr.w[4] = bfpack(kc.x, kc.y); bfr.w[5] = bfpack(kc.z, kc.w);
      bfr.w[6] = bfpack(kd.x, kd.y); bfr.w[7] = bfpack(kd.z, kd.w);

      sacc = __builtin_amdgcn_wmma_f32_16x16x32_bf16(false, af.v, false, bfr.v,
                                                     (short)0, sacc, false, false);
    }
    #pragma unroll
    for (int r = 0; r < 8; ++r) {
      const int qi = q0 + rb + r;
      const int kj = j0 + col;
      float s = sacc[r] * scale;
      if (MASKED) {
        const int d = qi - kj;
        // power-of-two dilation: branchless bitwise mask test
        const bool ok = (d >= 0) & ((d & dil_m1) == 0) & ((d >> dil_sh) < ks);
        s = ok ? s : -1.0e9f;
      }
      Slds[(rb + r) * Ntok + kj] = s;
    }
  }
  __syncthreads();

  // ---- phase 2: row softmax (2 lanes per row) ----
  const int half = lane >> 4;
  const int j_lo = half * (Ntok >> 1);
  const int j_hi = j_lo + (Ntok >> 1);
  float mx = -3.4e38f;
  for (int j = j_lo; j < j_hi; ++j) mx = fmaxf(mx, Slds[row * Ntok + j]);
  mx = fmaxf(mx, __shfl_xor(mx, 16, 32));
  float sum = 0.0f;
  for (int j = j_lo; j < j_hi; ++j) {
    float p = __expf(Slds[row * Ntok + j] - mx);
    Slds[row * Ntok + j] = p;
    sum += p;
  }
  sum += __shfl_xor(sum, 16, 32);
  if (lane < 16) rinv[lane] = 1.0f / sum;
  __syncthreads();

  // ---- phase 3: O = P V  (P fragments out of LDS via b128 loads) ----
  for (int d0 = 0; d0 < hd; d0 += 16) {
    v8f oacc = {};
    for (int k0 = 0; k0 < Ntok; k0 += 32) {
      Frag af, bfr;
      const float* prow = &Slds[row * Ntok + k0 + akb];
      const float4 pa = *(const float4*)(prow);
      const float4 pb = *(const float4*)(prow + 4);
      const float4 pc = *(const float4*)(prow + 16);
      const float4 pd = *(const float4*)(prow + 20);
      af.w[0] = bfpack(pa.x, pa.y); af.w[1] = bfpack(pa.z, pa.w);
      af.w[2] = bfpack(pb.x, pb.y); af.w[3] = bfpack(pb.z, pb.w);
      af.w[4] = bfpack(pc.x, pc.y); af.w[5] = bfpack(pc.z, pc.w);
      af.w[6] = bfpack(pd.x, pd.y); af.w[7] = bfpack(pd.z, pd.w);

      #pragma unroll
      for (int e = 0; e < 16; ++e)
        bfr.u[e] = bf16bits(Vp[(long long)(k0 + akb + e + (e & 8)) * rs + d0 + col]);

      oacc = __builtin_amdgcn_wmma_f32_16x16x32_bf16(false, af.v, false, bfr.v,
                                                     (short)0, oacc, false, false);
    }
    #pragma unroll
    for (int r = 0; r < 8; ++r) {
      const int qi = q0 + rb + r;
      O[((long long)b * Ntok + qi) * Dm + h * hd + d0 + col] = oacc[r] * rinv[rb + r];
    }
  }
}

// =====================================================================
// Row LayerNorm: rows of length n, one block (256 thr) per row.
// =====================================================================
__global__ void ln_kernel(const float* __restrict__ x, const float* __restrict__ g,
                          const float* __restrict__ bta, float* __restrict__ y, int n) {
  __shared__ float red[32];
  __shared__ float mean_s, rstd_s;
  const long long row = blockIdx.x;
  const float* xr = x + row * n;
  float*       yr = y + row * n;
  const int tid = threadIdx.x;
  const int nw  = blockDim.x >> 5;

  float s = 0.0f;
  for (int i = tid; i < n; i += blockDim.x) s += xr[i];
  for (int off = 16; off > 0; off >>= 1) s += __shfl_down(s, off, 32);
  if ((tid & 31) == 0) red[tid >> 5] = s;
  __syncthreads();
  if (tid < 32) {
    float t = (tid < nw) ? red[tid] : 0.0f;
    for (int off = 16; off > 0; off >>= 1) t += __shfl_down(t, off, 32);
    if (tid == 0) mean_s = t / (float)n;
  }
  __syncthreads();
  const float mean = mean_s;

  float v = 0.0f;
  for (int i = tid; i < n; i += blockDim.x) { float d = xr[i] - mean; v += d * d; }
  for (int off = 16; off > 0; off >>= 1) v += __shfl_down(v, off, 32);
  if ((tid & 31) == 0) red[tid >> 5] = v;
  __syncthreads();
  if (tid < 32) {
    float t = (tid < nw) ? red[tid] : 0.0f;
    for (int off = 16; off > 0; off >>= 1) t += __shfl_down(t, off, 32);
    if (tid == 0) rstd_s = rsqrtf(t / (float)n + 1e-5f);
  }
  __syncthreads();
  const float rstd = rstd_s;
  for (int i = tid; i < n; i += blockDim.x)
    yr[i] = (xr[i] - mean) * rstd * g[i] + bta[i];
}

// =====================================================================
// Batched 2D transpose: in [B,R,C] -> out [B,C,R]
// =====================================================================
__global__ void transpose_kernel(const float* __restrict__ X, float* __restrict__ Y,
                                 int R, int C) {
  __shared__ float tile[32][33];
  const int b = blockIdx.z;
  const int r0 = blockIdx.y * 32, c0 = blockIdx.x * 32;
  const float* Xb = X + (long long)b * R * C;
  float*       Yb = Y + (long long)b * R * C;
  const int tx = threadIdx.x, ty = threadIdx.y;
  for (int i = ty; i < 32; i += 8) {
    int r = r0 + i, c = c0 + tx;
    tile[i][tx] = (r < R && c < C) ? Xb[(long long)r * C + c] : 0.0f;
  }
  __syncthreads();
  for (int i = ty; i < 32; i += 8) {
    int c = c0 + i, r = r0 + tx;
    if (c < C && r < R) Yb[(long long)c * R + r] = tile[tx][i];
  }
}

// =====================================================================
// pos softmax: p[t] = softmax_t((t/T)*cw + cb), one block of T threads
// =====================================================================
__global__ void pos_softmax_kernel(const float* __restrict__ cw,
                                   const float* __restrict__ cb,
                                   float* __restrict__ p, int T) {
  __shared__ float red[32];
  __shared__ float mx_s, sum_s;
  const int t = threadIdx.x;
  const int nw = T >> 5;
  float v = ((float)t / (float)T) * cw[0] + cb[0];
  float m = v;
  for (int off = 16; off > 0; off >>= 1) m = fmaxf(m, __shfl_xor(m, off, 32));
  if ((t & 31) == 0) red[t >> 5] = m;
  __syncthreads();
  if (t < 32) {
    float mm = (t < nw) ? red[t] : -3.4e38f;
    for (int off = 16; off > 0; off >>= 1) mm = fmaxf(mm, __shfl_xor(mm, off, 32));
    if (t == 0) mx_s = mm;
  }
  __syncthreads();
  float e = __expf(v - mx_s);
  float s = e;
  for (int off = 16; off > 0; off >>= 1) s += __shfl_xor(s, off, 32);
  if ((t & 31) == 0) red[t >> 5] = s;
  __syncthreads();
  if (t < 32) {
    float ss = (t < nw) ? red[t] : 0.0f;
    for (int off = 16; off > 0; off >>= 1) ss += __shfl_xor(ss, off, 32);
    if (t == 0) sum_s = ss;
  }
  __syncthreads();
  p[t] = e / sum_s;
}

// =====================================================================
// Embedding: x_t [B,T,D] (last col = pos softmax) and x_s [B,D,T] (+PE)
// =====================================================================
__global__ void embed_kernel(const float* __restrict__ x,
                             const float* __restrict__ etw, const float* __restrict__ etb,
                             const float* __restrict__ esw, const float* __restrict__ esb,
                             const float* __restrict__ p,
                             float* __restrict__ xt, float* __restrict__ xs,
                             int B, int T, int D) {
  const int t = blockIdx.x, b = blockIdx.y, d = threadIdx.x;
  const float xv = x[(long long)b * T + t];
  float vt = (d < D - 1) ? (xv * etw[d] + etb[d]) : p[t];
  xt[((long long)b * T + t) * D + d] = vt;

  const int i2 = t >> 1;
  const float div = __expf((float)(2 * i2) * (-logf(10000.0f) / (float)T));
  const float ang = (float)d * div;
  const float pe  = (t & 1) ? __cosf(ang) : __sinf(ang);
  xs[((long long)b * D + d) * T + t] = xv * esw[d] + esb[d] + pe;
}

// =====================================================================
// last-step extraction + gate mix
// =====================================================================
__global__ void extract_last_t_kernel(const float* __restrict__ xt,
                                      float* __restrict__ out, int T, int D) {
  const int b = blockIdx.x, d = threadIdx.x;
  out[(long long)b * D + d] = xt[((long long)b * T + (T - 1)) * D + d];
}
__global__ void extract_last_s_kernel(const float* __restrict__ xs,
                                      float* __restrict__ out, int T, int D) {
  const int b = blockIdx.x, d = threadIdx.x;
  out[(long long)b * D + d] = xs[((long long)b * D + d) * T + (T - 1)];
}
__global__ void gate_mix_kernel(const float* __restrict__ g1, const float* __restrict__ g2,
                                const float* __restrict__ a, const float* __restrict__ c,
                                float* __restrict__ out, long long n) {
  const long long i = (long long)blockIdx.x * blockDim.x + threadIdx.x;
  if (i < n) {
    const float z = 1.0f / (1.0f + __expf(-(g1[i] + g2[i])));
    out[i] = z * a[i] + (1.0f - z) * c[i];
  }
}

// ---------------- host-side parameter structs ----------------
struct FFNP  { const float *ng, *nb, *w1, *b1, *w2, *b2; };
struct AttnP { const float *ng, *nb, *qkv, *pw, *pb; FFNP ffn; };
struct CafP  { const float *ngt, *nbt, *ngs, *nbs, *qkvt, *qkvs,
                           *pwt, *pbt, *pws, *pbs; FFNP ffnt, ffns; };
struct LayerP { AttnP dc[DEPc]; AttnP im; CafP caf; };
struct GFP   { const float *ngt, *nbt, *ngs, *nbs, *wt, *bt, *ws, *bs; };

extern "C" void kernel_launch(void* const* d_in, const int* in_sizes, int n_in,
                              void* d_out, int out_size, void* d_ws, size_t ws_size,
                              hipStream_t stream) {
  (void)in_sizes; (void)n_in; (void)out_size; (void)ws_size;
  // ---- walk params in setup_inputs() dict order ----
  int ip = 0;
  auto P = [&]() { return (const float*)d_in[ip++]; };
  const float* x       = P();
  const float* emb_t_w = P(); const float* emb_t_b = P();
  const float* emb_s_w = P(); const float* emb_s_b = P();
  const float* conv_w  = P(); const float* conv_b  = P();
  auto getF = [&](FFNP& f) { f.ng=P(); f.nb=P(); f.w1=P(); f.b1=P(); f.w2=P(); f.b2=P(); };
  LayerP L[NLc];
  for (int l = 0; l < NLc; ++l) {
    for (int d = 0; d < DEPc; ++d) {
      AttnP& a = L[l].dc[d];
      a.ng=P(); a.nb=P(); a.qkv=P(); a.pw=P(); a.pb=P(); getF(a.ffn);
    }
    AttnP& a = L[l].im;
    a.ng=P(); a.nb=P(); a.qkv=P(); a.pw=P(); a.pb=P(); getF(a.ffn);
    CafP& c = L[l].caf;
    c.ngt=P(); c.nbt=P(); c.ngs=P(); c.nbs=P();
    c.qkvt=P(); c.qkvs=P(); c.pwt=P(); c.pbt=P(); c.pws=P(); c.pbs=P();
    getF(c.ffnt); getF(c.ffns);
  }
  GFP gf;
  gf.ngt=P(); gf.nbt=P(); gf.ngs=P(); gf.nbs=P();
  gf.wt=P();  gf.bt=P();  gf.ws=P();  gf.bs=P();
  FFNP soh, rul; getF(soh); getF(rul);

  // ---- workspace carve-out (floats) ----
  float* ws = (float*)d_ws;
  size_t off = 0;
  auto alloc = [&](size_t n) { float* p = ws + off; off += n; return p; };
  const size_t NTD = (size_t)Bc * Tc * Dc;
  float* XT   = alloc(NTD);                 // x_t  [B,T,D]
  float* XS   = alloc(NTD);                 // x_s  [B,D,T]
  float* Hb   = alloc(NTD);                 // LN scratch / 2nd attn out
  float* QKV1 = alloc((size_t)Bc * Tc * 3 * Dc);
  float* QKV2 = alloc((size_t)Bc * Tc * 3 * Dc);
  float* AO   = alloc(NTD);                 // attention output
  float* FH   = alloc((size_t)Bc * Tc * 4 * Dc); // FFN hidden
  float* TMP  = alloc(NTD);
  float* Pbuf = alloc(Tc);
  float* LT   = alloc((size_t)Bc * Dc);
  float* LS   = alloc((size_t)Bc * Dc);
  float* LTn  = alloc((size_t)Bc * Dc);
  float* LSn  = alloc((size_t)Bc * Dc);
  float* G1   = alloc((size_t)Bc * Dc);
  float* G2   = alloc((size_t)Bc * Dc);
  float* FUSED= alloc((size_t)Bc * Dc);
  float* SH   = alloc((size_t)Bc * Dc);
  float* HH   = alloc((size_t)Bc * 4 * Dc);

  // ---- launch helpers ----
  auto gemm = [&](const float* A, int lda, long long sA, const float* W,
                  const float* bias, const float* resid, int ldr, long long sR,
                  float* C, int ldc, long long sC,
                  int batch, int M, int N, int K, int act) {
    dim3 g((N + 127) / 128, (M + 15) / 16, batch);
    const bool guard = (M & 15) || (N & 127);
    if (guard)
      wmma_gemm_kernel<true><<<g, 128, 0, stream>>>(A, lda, sA, W, bias, resid, ldr, sR,
                                                    C, ldc, sC, M, N, K, act);
    else
      wmma_gemm_kernel<false><<<g, 128, 0, stream>>>(A, lda, sA, W, bias, resid, ldr, sR,
                                                     C, ldc, sC, M, N, K, act);
  };
  auto ln = [&](const float* X, const float* g, const float* b, float* Y,
                int rows, int n) {
    ln_kernel<<<rows, 256, 0, stream>>>(X, g, b, Y, n);
  };
  // dil_log2 < 0  -> unmasked
  auto attn = [&](const float* Qb, const float* KVb, float* O,
                  int Ntok, int heads, int hd, float scale, int dil_log2) {
    dim3 g(Ntok / 16, heads, Bc);
    size_t sm = (size_t)16 * Ntok * sizeof(float);
    if (dil_log2 >= 0)
      wmma_attn_kernel<true><<<g, 32, sm, stream>>>(
          Qb, KVb, O, Ntok, heads, hd, scale,
          (1 << dil_log2) - 1, dil_log2, KSc);
    else
      wmma_attn_kernel<false><<<g, 32, sm, stream>>>(
          Qb, KVb, O, Ntok, heads, hd, scale, 0, 0, KSc);
  };
  auto transpose = [&](const float* X, float* Y, int R, int C) {
    dim3 g((C + 31) / 32, (R + 31) / 32, Bc);
    transpose_kernel<<<g, dim3(32, 8), 0, stream>>>(X, Y, R, C);
  };

  const float scale_t = 1.0f / sqrtf((float)(Dc / Hc));   // 32^-0.5
  const float scale_s = 1.0f / sqrtf((float)(Tc / HSc));  // 64^-0.5

  // ---- embedding ----
  pos_softmax_kernel<<<1, Tc, 0, stream>>>(conv_w, conv_b, Pbuf, Tc);
  embed_kernel<<<dim3(Tc, Bc), Dc, 0, stream>>>(x, emb_t_w, emb_t_b, emb_s_w, emb_s_b,
                                                Pbuf, XT, XS, Bc, Tc, Dc);

  // ---- layers ----
  for (int l = 0; l < NLc; ++l) {
    // ----- 3x dilated-causal multi-head self-attention on x_t -----
    for (int d = 0; d < DEPc; ++d) {
      const AttnP& a = L[l].dc[d];
      ln(XT, a.ng, a.nb, Hb, Bc * Tc, Dc);
      gemm(Hb, Dc, (long long)Tc * Dc, a.qkv, nullptr, nullptr, 0, 0,
           QKV1, 3 * Dc, (long long)Tc * 3 * Dc, Bc, Tc, 3 * Dc, Dc, 0);
      attn(QKV1, QKV1, AO, Tc, Hc, Dc / Hc, scale_t, d);   // dilation 2^d
      gemm(AO, Dc, (long long)Tc * Dc, a.pw, a.pb, XT, Dc, (long long)Tc * Dc,
           XT, Dc, (long long)Tc * Dc, Bc, Tc, Dc, Dc, 0);
      // FFN with residual
      ln(XT, a.ffn.ng, a.ffn.nb, Hb, Bc * Tc, Dc);
      gemm(Hb, Dc, (long long)Tc * Dc, a.ffn.w1, a.ffn.b1, nullptr, 0, 0,
           FH, 4 * Dc, (long long)Tc * 4 * Dc, Bc, Tc, 4 * Dc, Dc, 1);
      gemm(FH, 4 * Dc, (long long)Tc * 4 * Dc, a.ffn.w2, a.ffn.b2,
           XT, Dc, (long long)Tc * Dc, XT, Dc, (long long)Tc * Dc,
           Bc, Tc, Dc, 4 * Dc, 0);
    }

    // ----- spectral self-attention on x_s ([B,D,T]: D tokens of dim T) -----
    {
      const AttnP& a = L[l].im;
      ln(XS, a.ng, a.nb, Hb, Bc * Dc, Tc);
      gemm(Hb, Tc, (long long)Dc * Tc, a.qkv, nullptr, nullptr, 0, 0,
           QKV1, 3 * Tc, (long long)Dc * 3 * Tc, Bc, Dc, 3 * Tc, Tc, 0);
      attn(QKV1, QKV1, AO, Dc, HSc, Tc / HSc, scale_s, -1);
      gemm(AO, Tc, (long long)Dc * Tc, a.pw, a.pb, XS, Tc, (long long)Dc * Tc,
           XS, Tc, (long long)Dc * Tc, Bc, Dc, Tc, Tc, 0);
      ln(XS, a.ffn.ng, a.ffn.nb, Hb, Bc * Dc, Tc);
      gemm(Hb, Tc, (long long)Dc * Tc, a.ffn.w1, a.ffn.b1, nullptr, 0, 0,
           FH, 4 * Tc, (long long)Dc * 4 * Tc, Bc, Dc, 4 * Tc, Tc, 1);
      gemm(FH, 4 * Tc, (long long)Dc * 4 * Tc, a.ffn.w2, a.ffn.b2,
           XS, Tc, (long long)Dc * Tc, XS, Tc, (long long)Dc * Tc,
           Bc, Dc, Tc, 4 * Tc, 0);
    }

    // ----- cross-attention fusion -----
    {
      const CafP& c = L[l].caf;
      ln(XT, c.ngt, c.nbt, Hb, Bc * Tc, Dc);            // xt = LN(x_t)
      transpose(XS, TMP, Dc, Tc);                       // x_s^T -> [B,T,D]
      ln(TMP, c.ngs, c.nbs, TMP, Bc * Tc, Dc);          // xs = LN(x_s^T)
      gemm(Hb, Dc, (long long)Tc * Dc, c.qkvt, nullptr, nullptr, 0, 0,
           QKV1, 3 * Dc, (long long)Tc * 3 * Dc, Bc, Tc, 3 * Dc, Dc, 0);
      gemm(TMP, Dc, (long long)Tc * Dc, c.qkvs, nullptr, nullptr, 0, 0,
           QKV2, 3 * Dc, (long long)Tc * 3 * Dc, Bc, Tc, 3 * Dc, Dc, 0);
      attn(QKV2, QKV1, AO, Tc, Hc, Dc / Hc, scale_t, -1);  // ot = attend(qs, kt, vt)
      attn(QKV1, QKV2, Hb, Tc, Hc, Dc / Hc, scale_t, -1);  // os = attend(qt, ks, vs)

      // x_t = FFN(ot @ pwt + pbt)   (no residual)
      gemm(AO, Dc, (long long)Tc * Dc, c.pwt, c.pbt, nullptr, 0, 0,
           TMP, Dc, (long long)Tc * Dc, Bc, Tc, Dc, Dc, 0);
      ln(TMP, c.ffnt.ng, c.ffnt.nb, AO, Bc * Tc, Dc);
      gemm(AO, Dc, (long long)Tc * Dc, c.ffnt.w1, c.ffnt.b1, nullptr, 0, 0,
           FH, 4 * Dc, (long long)Tc * 4 * Dc, Bc, Tc, 4 * Dc, Dc, 1);
      gemm(FH, 4 * Dc, (long long)Tc * 4 * Dc, c.ffnt.w2, c.ffnt.b2, nullptr, 0, 0,
           XT, Dc, (long long)Tc * Dc, Bc, Tc, Dc, 4 * Dc, 0);

      // x_s = FFN(os @ pws + pbs).transpose
      gemm(Hb, Dc, (long long)Tc * Dc, c.pws, c.pbs, nullptr, 0, 0,
           TMP, Dc, (long long)Tc * Dc, Bc, Tc, Dc, Dc, 0);
      ln(TMP, c.ffns.ng, c.ffns.nb, Hb, Bc * Tc, Dc);
      gemm(Hb, Dc, (long long)Tc * Dc, c.ffns.w1, c.ffns.b1, nullptr, 0, 0,
           FH, 4 * Dc, (long long)Tc * 4 * Dc, Bc, Tc, 4 * Dc, Dc, 1);
      gemm(FH, 4 * Dc, (long long)Tc * 4 * Dc, c.ffns.w2, c.ffns.b2, nullptr, 0, 0,
           TMP, Dc, (long long)Tc * Dc, Bc, Tc, Dc, 4 * Dc, 0);
      transpose(TMP, XS, Tc, Dc);                       // back to [B,D,T]
    }
  }

  // ---- gated fusion + heads (only last time step is needed) ----
  extract_last_t_kernel<<<Bc, Dc, 0, stream>>>(XT, LT, Tc, Dc);
  extract_last_s_kernel<<<Bc, Dc, 0, stream>>>(XS, LS, Tc, Dc);
  ln(LT, gf.ngt, gf.nbt, LTn, Bc, Dc);
  ln(LS, gf.ngs, gf.nbs, LSn, Bc, Dc);
  gemm(LTn, Dc, 0, gf.wt, gf.bt, nullptr, 0, 0, G1, Dc, 0, 1, Bc, Dc, Dc, 0);
  gemm(LSn, Dc, 0, gf.ws, gf.bs, nullptr, 0, 0, G2, Dc, 0, 1, Bc, Dc, Dc, 0);
  gate_mix_kernel<<<(Bc * Dc + 255) / 256, 256, 0, stream>>>(
      G1, G2, LTn, LSn, FUSED, (long long)Bc * Dc);

  float* out = (float*)d_out;                 // [B, PRED+1]
  const int ldo = PREDc + 1;
  // soh head
  ln(FUSED, soh.ng, soh.nb, SH, Bc, Dc);
  gemm(SH, Dc, 0, soh.w1, soh.b1, nullptr, 0, 0, HH, 4 * Dc, 0, 1, Bc, 4 * Dc, Dc, 1);
  gemm(HH, 4 * Dc, 0, soh.w2, soh.b2, nullptr, 0, 0, out, ldo, 0, 1, Bc, PREDc, 4 * Dc, 0);
  // rul head
  ln(FUSED, rul.ng, rul.nb, SH, Bc, Dc);
  gemm(SH, Dc, 0, rul.w1, rul.b1, nullptr, 0, 0, HH, 4 * Dc, 0, 1, Bc, 4 * Dc, Dc, 1);
  gemm(HH, 4 * Dc, 0, rul.w2, rul.b2, nullptr, 0, 0, out + PREDc, ldo, 0, 1, Bc, 1, 4 * Dc, 0);
}
